// BayesianBeliefNetwork_20229295964734
// MI455X (gfx1250) — compile-verified, matrix operand order checked by reading, not run on hardware
//
#include <hip/hip_runtime.h>
#include <hip/hip_bf16.h>

typedef __attribute__((ext_vector_type(8)))  float  v8f;
typedef __attribute__((ext_vector_type(16))) __bf16 v16bf;
typedef __attribute__((ext_vector_type(8)))  __bf16 v8bf;

#define HID   2048
#define NVARS 10
#define D4    512
#define BATCH 4
#define SEQ   2048
#define NTOK  (BATCH * SEQ)          // 8192
#define COMB  (HID + NVARS)          // 2058

// ---- GEMM tiling ----
#define BM 128
#define BN 128
#define BK 32
#define LDP 40   // padded LDS K-stride (bf16 elems): 80B rows, conflict-free b128 loads
#define NSTAGES (HID / BK)           // 64

// ---------------------------------------------------------------------------
// CDNA5 async global->LDS helpers
// ---------------------------------------------------------------------------
__device__ __forceinline__ unsigned lds_addr32(const void* p) {
    // generic LDS pointers carry the wave-relative LDS byte offset in addr[31:0]
    return (unsigned)(unsigned long long)p;
}
__device__ __forceinline__ void async_copy_b128(unsigned lds, const void* g) {
    asm volatile("global_load_async_to_lds_b128 %0, %1, off"
                 :: "v"(lds), "v"((unsigned long long)g) : "memory");
}
__device__ __forceinline__ void wait_async_le4() {
    asm volatile("s_wait_asynccnt 0x4" ::: "memory");
}
__device__ __forceinline__ void wait_async_0() {
    asm volatile("s_wait_asynccnt 0x0" ::: "memory");
}

// ---------------------------------------------------------------------------
// Kernel 0: zero the evidence accumulator (40 floats)
// ---------------------------------------------------------------------------
__global__ void init_acc_kernel(float* acc) {
    if (threadIdx.x < BATCH * NVARS) acc[threadIdx.x] = 0.0f;
}

// ---------------------------------------------------------------------------
// Kernel 1: per-token LayerNorm -> bf16 h, plus partial evidence logits
//           acc[b,v] += sum_i h[token,i] * W_ve[v,i]   (atomic over tokens)
// ---------------------------------------------------------------------------
__global__ __launch_bounds__(256)
void ln_kernel(const float* __restrict__ x, const float* __restrict__ gamma,
               const float* __restrict__ beta, const float* __restrict__ Wve,
               __bf16* __restrict__ hb, float* __restrict__ acc) {
    const int m = blockIdx.x;              // token index
    const int t = threadIdx.x;
    const float* xr = x + (size_t)m * HID;

    float v[8];
    float s = 0.0f, ss = 0.0f;
    #pragma unroll
    for (int j = 0; j < 8; ++j) {
        float f = xr[t + j * 256];
        v[j] = f; s += f; ss += f * f;
    }

    __shared__ float red[256];
    __shared__ float s_mu, s_rstd;

    red[t] = s; __syncthreads();
    for (int off = 128; off > 0; off >>= 1) {
        if (t < off) red[t] += red[t + off];
        __syncthreads();
    }
    if (t == 0) s_mu = red[0] * (1.0f / HID);
    __syncthreads();

    red[t] = ss; __syncthreads();
    for (int off = 128; off > 0; off >>= 1) {
        if (t < off) red[t] += red[t + off];
        __syncthreads();
    }
    if (t == 0) {
        float var = red[0] * (1.0f / HID) - s_mu * s_mu;   // biased variance
        s_rstd = rsqrtf(var + 1e-5f);
    }
    __syncthreads();

    const float mu = s_mu, rstd = s_rstd;
    float part[NVARS];
    #pragma unroll
    for (int vv = 0; vv < NVARS; ++vv) part[vv] = 0.0f;

    #pragma unroll
    for (int j = 0; j < 8; ++j) {
        int i = t + j * 256;
        float y = (v[j] - mu) * rstd * gamma[i] + beta[i];
        hb[(size_t)m * HID + i] = (__bf16)y;
        #pragma unroll
        for (int vv = 0; vv < NVARS; ++vv)
            part[vv] += y * Wve[vv * HID + i];
    }

    const int b = m >> 11;                 // m / SEQ
    for (int vv = 0; vv < NVARS; ++vv) {
        red[t] = part[vv]; __syncthreads();
        for (int off = 128; off > 0; off >>= 1) {
            if (t < off) red[t] += red[t + off];
            __syncthreads();
        }
        if (t == 0) atomicAdd(&acc[b * NVARS + vv], red[0]);
        __syncthreads();
    }
}

// ---------------------------------------------------------------------------
// Kernel 2: convert W_out[:, :HID] (row-major, stride COMB) -> bf16 [HID,HID]
// ---------------------------------------------------------------------------
__global__ __launch_bounds__(256)
void conv_wout_kernel(const float* __restrict__ W, __bf16* __restrict__ Wb) {
    int idx = blockIdx.x * 256 + threadIdx.x;   // over HID*HID
    int n = idx >> 11, k = idx & (HID - 1);
    Wb[idx] = (__bf16)W[(size_t)n * COMB + k];
}

// ---------------------------------------------------------------------------
// Kernel 3: evidence -> 5-iteration belief propagation -> posterior,
//           then postc[b,n] = b_out[n] + sum_v posterior[b,v]*W_out[n, HID+v]
// ---------------------------------------------------------------------------
__global__ __launch_bounds__(256)
void bp_kernel(const float* __restrict__ acc, const float* __restrict__ bve,
               const int* __restrict__ parents, const float* __restrict__ var_emb,
               const float* __restrict__ cpt_emb, const float* __restrict__ W_out,
               const float* __restrict__ b_out, float* __restrict__ postc) {
    __shared__ float probs[BATCH][NVARS], newp[BATCH][NVARS];
    __shared__ float cptn[NVARS], denom[NVARS];
    __shared__ int   pf[NVARS][NVARS], cnt[NVARS];
    __shared__ unsigned char updm[BATCH][NVARS];
    __shared__ float red[256], red2[256];

    const int t = threadIdx.x;
    if (t < NVARS * NVARS) pf[t / NVARS][t % NVARS] = parents[t];
    __syncthreads();
    if (t < NVARS) {
        int c = 0;
        for (int p = 0; p < NVARS; ++p) c += pf[t][p];
        cnt[t] = c;
        denom[t] = (c > 0) ? (float)c : 1.0f;
    }
    if (t < BATCH * NVARS) {
        int vv = t % NVARS;
        float e = 1.0f / (1.0f + expf(-(acc[t] * (1.0f / SEQ) + bve[vv])));
        probs[t / NVARS][vv] = e;
    }
    __syncthreads();
    if (t < BATCH * NVARS) {
        int b = t / NVARS, vv = t % NVARS;
        float e = probs[b][vv];
        updm[b][vv] = (e > 0.1f) && (e < 0.9f) && (cnt[vv] > 0);
    }
    // cpt norms
    for (int vv = 0; vv < NVARS; ++vv) {
        float p = 0.0f;
        for (int d = t; d < D4; d += 256) {
            float c = cpt_emb[vv * D4 + d];
            p += c * c;
        }
        red[t] = p; __syncthreads();
        for (int off = 128; off > 0; off >>= 1) {
            if (t < off) red[t] += red[t + off];
            __syncthreads();
        }
        if (t == 0) cptn[vv] = fmaxf(sqrtf(red[0]), 1e-8f);
        __syncthreads();
    }
    // iterations
    for (int iter = 0; iter < 5; ++iter) {
        for (int b = 0; b < BATCH; ++b) {
            for (int vv = 0; vv < NVARS; ++vv) {
                float dotp = 0.0f, sqp = 0.0f;
                for (int d = t; d < D4; d += 256) {
                    float pe = 0.0f;
                    for (int p = 0; p < NVARS; ++p)
                        if (pf[vv][p]) pe += probs[b][p] * var_emb[p * D4 + d];
                    pe /= denom[vv];
                    dotp += pe * cpt_emb[vv * D4 + d];
                    sqp  += pe * pe;
                }
                red[t] = dotp; red2[t] = sqp; __syncthreads();
                for (int off = 128; off > 0; off >>= 1) {
                    if (t < off) { red[t] += red[t + off]; red2[t] += red2[t + off]; }
                    __syncthreads();
                }
                if (t == 0) {
                    float pen  = fmaxf(sqrtf(red2[0]), 1e-8f);
                    float cond = 1.0f / (1.0f + expf(-(red[0] / (pen * cptn[vv]))));
                    newp[b][vv] = updm[b][vv] ? cond : probs[b][vv];
                }
                __syncthreads();
            }
        }
        if (t < BATCH * NVARS) probs[t / NVARS][t % NVARS] = newp[t / NVARS][t % NVARS];
        __syncthreads();
    }
    // posterior contribution + output bias, folded per (b, n)
    for (int n = t; n < HID; n += 256) {
        float bo = b_out[n];
        const float* wrow = W_out + (size_t)n * COMB + HID;
        for (int b = 0; b < BATCH; ++b) {
            float pc = bo;
            #pragma unroll
            for (int vv = 0; vv < NVARS; ++vv) pc += probs[b][vv] * wrow[vv];
            postc[b * HID + n] = pc;
        }
    }
}

// ---------------------------------------------------------------------------
// Kernel 4: main GEMM  out[m,n] = sum_k hb[m,k]*Wb[n,k] + residual + postc
//           bf16 WMMA 16x16x32, f32 accumulate, async double-buffered LDS
// ---------------------------------------------------------------------------
__device__ __forceinline__ v16bf load_frag(const __bf16* base, int lane) {
    // 16-bit A/B fragment per ISA: lanes 0-15 -> K {0..7,16..23}, lanes 16-31 -> K {8..15,24..31}
    int r  = lane & 15;
    int kh = (lane >> 4) << 3;                 // 0 or 8
    const __bf16* p = base + r * LDP + kh;
    union { v16bf v; v8bf h[2]; } u;
    u.h[0] = *(const v8bf*)(p);                // K = kh + 0..7
    u.h[1] = *(const v8bf*)(p + 16);           // K = kh + 16..23
    return u.v;
}

__global__ __launch_bounds__(256)
void gemm_kernel(const __bf16* __restrict__ hb, const __bf16* __restrict__ Wb,
                 const float* __restrict__ residual, const float* __restrict__ postc,
                 float* __restrict__ out) {
    __shared__ __align__(16) __bf16 As[2][BM * LDP];
    __shared__ __align__(16) __bf16 Bs[2][BN * LDP];

    const int tid   = threadIdx.x;
    const int lane  = tid & 31;
    const int w     = tid >> 5;
    const int waveM = w & 1;                   // 2 wave rows
    const int waveN = w >> 1;                  // 4 wave cols
    const int bM = (blockIdx.x & 63) * BM;     // 64 M-tiles
    const int bN = (blockIdx.x >> 6) * BN;     // 16 N-tiles

    // this thread's two 16B chunks of the 128x32 tile (512 chunks total)
    const int idx0 = tid,        idx1 = tid + 256;
    const int row0 = idx0 >> 2,  kc0 = (idx0 & 3) << 3;
    const int row1 = idx1 >> 2,  kc1 = (idx1 & 3) << 3;

    // per-thread global source pointers, advanced by BK per stage
    const __bf16* gA0 = hb + (size_t)(bM + row0) * HID + kc0;
    const __bf16* gA1 = hb + (size_t)(bM + row1) * HID + kc1;
    const __bf16* gB0 = Wb + (size_t)(bN + row0) * HID + kc0;
    const __bf16* gB1 = Wb + (size_t)(bN + row1) * HID + kc1;

    // per-thread LDS destinations (byte offsets), per buffer
    unsigned lA0[2], lA1[2], lB0[2], lB1[2];
    #pragma unroll
    for (int bi = 0; bi < 2; ++bi) {
        lA0[bi] = lds_addr32(&As[bi][row0 * LDP + kc0]);
        lA1[bi] = lds_addr32(&As[bi][row1 * LDP + kc1]);
        lB0[bi] = lds_addr32(&Bs[bi][row0 * LDP + kc0]);
        lB1[bi] = lds_addr32(&Bs[bi][row1 * LDP + kc1]);
    }

    v8f acc[4][2];
    #pragma unroll
    for (int mt = 0; mt < 4; ++mt)
        #pragma unroll
        for (int nt = 0; nt < 2; ++nt)
            #pragma unroll
            for (int i = 0; i < 8; ++i) acc[mt][nt][i] = 0.0f;

    // prologue: stage 0 into buffer 0
    async_copy_b128(lA0[0], gA0);
    async_copy_b128(lB0[0], gB0);
    async_copy_b128(lA1[0], gA1);
    async_copy_b128(lB1[0], gB1);

    for (int s = 0; s < NSTAGES; ++s) {
        const int cur = s & 1;
        if (s + 1 < NSTAGES) {
            const int nxt = cur ^ 1;
            const int koff = (s + 1) * BK;
            async_copy_b128(lA0[nxt], gA0 + koff);
            async_copy_b128(lB0[nxt], gB0 + koff);
            async_copy_b128(lA1[nxt], gA1 + koff);
            async_copy_b128(lB1[nxt], gB1 + koff);
            wait_async_le4();                  // stage-s loads complete (in order)
        } else {
            wait_async_0();
        }
        __syncthreads();                       // all waves' stage-s tile visible

        v16bf bfr[2];
        #pragma unroll
        for (int nt = 0; nt < 2; ++nt)
            bfr[nt] = load_frag(&Bs[cur][(waveN * 32 + nt * 16) * LDP], lane);
        #pragma unroll
        for (int mt = 0; mt < 4; ++mt) {
            v16bf af = load_frag(&As[cur][(waveM * 64 + mt * 16) * LDP], lane);
            #pragma unroll
            for (int nt = 0; nt < 2; ++nt)
                acc[mt][nt] = __builtin_amdgcn_wmma_f32_16x16x32_bf16(
                    false, af, false, bfr[nt], (short)0, acc[mt][nt], false, false);
        }
        __syncthreads();                       // done reading buf[cur] before stage s+2 overwrites
    }

    // epilogue: D layout -> lane%16 = column, VGPR r = row r (+8 for upper half-wave)
    const int col = lane & 15;
    const int rhi = (lane >> 4) << 3;
    #pragma unroll
    for (int mt = 0; mt < 4; ++mt) {
        #pragma unroll
        for (int nt = 0; nt < 2; ++nt) {
            int n = bN + waveN * 32 + nt * 16 + col;
            #pragma unroll
            for (int r = 0; r < 8; ++r) {
                int m = bM + waveM * 64 + mt * 16 + rhi + r;
                int b = m >> 11;
                size_t o = (size_t)m * HID + n;
                out[o] = acc[mt][nt][r] + residual[o] + postc[b * HID + n];
            }
        }
    }
}

// ---------------------------------------------------------------------------
extern "C" void kernel_launch(void* const* d_in, const int* in_sizes, int n_in,
                              void* d_out, int out_size, void* d_ws, size_t ws_size,
                              hipStream_t stream) {
    const float* x       = (const float*)d_in[0];   // hidden_states [4,2048,2048]
    const float* gamma   = (const float*)d_in[1];
    const float* beta    = (const float*)d_in[2];
    const float* Wve     = (const float*)d_in[3];   // [10,2048]
    const float* bve     = (const float*)d_in[4];   // [10]
    const float* var_emb = (const float*)d_in[5];   // [10,512]
    const float* cpt_emb = (const float*)d_in[6];   // [10,512]
    const float* W_out   = (const float*)d_in[7];   // [2048,2058]
    const float* b_out   = (const float*)d_in[8];   // [2048]
    const int*   parents = (const int*)d_in[9];     // [10,10]
    float* out = (float*)d_out;

    char* ws = (char*)d_ws;
    const size_t hb_bytes = (size_t)NTOK * HID * 2;          // 32 MiB
    const size_t wb_bytes = (size_t)HID * HID * 2;           //  8 MiB
    __bf16* hb   = (__bf16*)(ws);
    __bf16* Wb   = (__bf16*)(ws + hb_bytes);
    float*  accp = (float*)(ws + hb_bytes + wb_bytes);       // 40 floats
    float*  postc = (float*)(ws + hb_bytes + wb_bytes + 256); // [4,2048]

    init_acc_kernel<<<1, 64, 0, stream>>>(accp);
    ln_kernel<<<NTOK, 256, 0, stream>>>(x, gamma, beta, Wve, hb, accp);
    conv_wout_kernel<<<(HID * HID) / 256, 256, 0, stream>>>(W_out, Wb);
    bp_kernel<<<1, 256, 0, stream>>>(accp, bve, parents, var_emb, cpt_emb, W_out, b_out, postc);
    gemm_kernel<<<64 * 16, 256, 0, stream>>>(hb, Wb, x, postc, out);
}
